// DeformShift_37580963840361
// MI455X (gfx1250) — matrix operand order repaired and checked by reading, not compiled, not run on hardware
//
#include <hip/hip_runtime.h>
#include <hip/hip_bf16.h>
#include <cmath>

// Problem constants (match reference)
#define BATCH 8
#define CH    192
#define HH    56
#define WW    56
#define HWSZ  (HH * WW)          // 3136
#define KTAP  9                  // 3x3 deform taps
#define OCH   18                 // 2*KK offset channels
#define KRED  (CH * KTAP)        // 1728 im2col reduction
#define MTOT  (BATCH * HWSZ)     // 25088 spatial positions (divisible by 16)

typedef __attribute__((ext_vector_type(16))) _Float16 v16h;
typedef __attribute__((ext_vector_type(8)))  float    v8f;

// ---------------------------------------------------------------------------
// Kernel 1: offset conv as WMMA GEMM, tap-major loop nest, ROLLED loops.
//   D[m][n] = sum_{t=0..8} sum_{c=0..191} A_t[m][c] * B_t[c][n]
//   A_t[m][c] = x[b, c, h+ky, w+kx]  (shifted plane, zero outside)
//   B_t[c][n] = w_off[n*1728 + c*9 + t]
// Loops kept rolled (#pragma unroll 1) so the live set stays ~80 VGPRs:
// no scratch spills, no s_set_vgpr_msb traffic. Fragment elements are loads
// at compile-time-constant immediate offsets off a per-tap base pointer.
// One wave per 16-pixel M tile; two f32 accumulators cover N=0..31 (18 live).
// ---------------------------------------------------------------------------
__global__ __launch_bounds__(128) void
deform_offset_wmma(const float* __restrict__ x,
                   const float* __restrict__ w_off,
                   const float* __restrict__ b_off,
                   float* __restrict__ off_buf)
{
    const int lane   = threadIdx.x & 31;
    const int wave   = threadIdx.x >> 5;
    const int tile   = blockIdx.x * 4 + wave;          // 0..1567
    const int laneLo = lane & 15;
    const int laneHi = lane >> 4;                      // 0 or 1

    // A-row (pixel) for this lane: row M = laneLo
    const int pix  = tile * 16 + laneLo;
    const int bimg = pix / HWSZ;
    const int hw   = pix - bimg * HWSZ;
    const int h    = hw / WW;
    const int w    = hw - h * WW;

    // B columns: N = laneLo (always < 18) and 16+laneLo (live for laneLo < 2)
    const int  n0  = laneLo;
    const int  n1  = 16 + laneLo;
    const bool n1v = (n1 < OCH);
    const int  n1c = n1v ? n1 : (OCH - 1);

    v8f acc0 = {};
    v8f acc1 = {};

    const float* ximg = x + (size_t)bimg * CH * HWSZ;

#pragma unroll 1
    for (int t = 0; t < KTAP; ++t) {
        const int ty = t / 3;
        const int ky = ty - 1;
        const int kx = (t - 3 * ty) - 1;
        const int yy = h + ky;
        const int xx = w + kx;
        const bool valid = (yy >= 0) && (yy < HH) && (xx >= 0) && (xx < WW);
        const int yyc = min(max(yy, 0), HH - 1);       // clamped -> safe address
        const int xxc = min(max(xx, 0), WW - 1);

        // Per-lane bases; the 16*laneHi channel shift of the chosen WMMA
        // register layout is folded into the base pointer once per tap.
        const float* ap  = ximg + (size_t)(16 * laneHi) * HWSZ + yyc * WW + xxc;
        const float* bp0 = w_off + (size_t)n0  * KRED + (16 * laneHi) * KTAP + t;
        const float* bp1 = w_off + (size_t)n1c * KRED + (16 * laneHi) * KTAP + t;

#pragma unroll 1
        for (int kb = 0; kb < CH; kb += 32) {
            v16h a, b0, b1;
#pragma unroll
            for (int j = 0; j < 16; ++j) {
                // A half j: K = 16*laneHi + j (folded)
                const int co = j;
                const float av = ap[(size_t)co * HWSZ];    // imm offset co*HW*4
                a[j] = (_Float16)(valid ? av : 0.0f);
                // B half j: K = 16*laneHi + j (folded), column = lane%16
                const float bv0 = bp0[(size_t)j * KTAP];   // imm offset j*36
                const float bv1 = bp1[(size_t)j * KTAP];
                b0[j] = (_Float16)bv0;
                b1[j] = (_Float16)(n1v ? bv1 : 0.0f);
            }

            // (neg_a, A, neg_b, B, c_mod, C, reuse_a, reuse_b)
            acc0 = __builtin_amdgcn_wmma_f32_16x16x32_f16(
                       false, a, false, b0, (short)0, acc0, false, false);
            acc1 = __builtin_amdgcn_wmma_f32_16x16x32_f16(
                       false, a, false, b1, (short)0, acc1, false, false);

            ap  += (size_t)32 * HWSZ;   // next 32 channels
            bp0 += 32 * KTAP;
            bp1 += 32 * KTAP;
        }
    }

    // ---- D layout: VGPR v holds M = v + 8*laneHi, N = lane%16 ----
    const float bias0 = b_off[n0];
    const float bias1 = n1v ? b_off[n1c] : 0.0f;
#pragma unroll
    for (int v = 0; v < 8; ++v) {
        const int m    = v + 8 * laneHi;
        const int pix2 = tile * 16 + m;
        const int bi   = pix2 / HWSZ;
        const int hw2  = pix2 - bi * HWSZ;
        off_buf[((size_t)bi * OCH + n0) * HWSZ + hw2] = acc0[v] + bias0;
        if (n1v)
            off_buf[((size_t)bi * OCH + n1) * HWSZ + hw2] = acc1[v] + bias1;
    }
}

// ---------------------------------------------------------------------------
// Kernel 2: fused bilinear deform-sample + depthwise 9-tap reduce.
// Block = (b, 64-pixel tile). Tap geometry (clamped corner indices + validity-
// folded bilinear weights) is computed ONCE in LDS and reused by all 192
// channels (offset_groups == 1). 256 threads = 64 pixels x 4 channels/iter.
// ---------------------------------------------------------------------------
__global__ __launch_bounds__(256) void
deform_apply(const float* __restrict__ x,
             const float* __restrict__ off_buf,
             const float* __restrict__ w_def,
             float* __restrict__ out)
{
    __shared__ int4   s_idx[KTAP * 64];   // clamped corner linear indices
    __shared__ float4 s_w[KTAP * 64];     // bilinear weights (zero if OOB)
    __shared__ float  s_wd[CH * KTAP];    // depthwise weights

    const int b        = blockIdx.y;
    const int tileBase = blockIdx.x * 64; // 49 tiles of 64 pixels = 3136
    const int tid      = threadIdx.x;

    // Preload depthwise weights to LDS
    for (int i = tid; i < CH * KTAP; i += 256) s_wd[i] = w_def[i];

    // Build tap tables (shared by all channels)
    for (int i = tid; i < KTAP * 64; i += 256) {
        const int k   = i >> 6;           // tap 0..8
        const int p   = i & 63;           // pixel in tile
        const int pix = tileBase + p;
        const int h   = pix / WW;
        const int w   = pix - h * WW;

        const float dy = off_buf[((size_t)b * OCH + 2 * k)     * HWSZ + pix];
        const float dx = off_buf[((size_t)b * OCH + 2 * k + 1) * HWSZ + pix];

        const float py = (float)h + (float)(k / 3 - 1) + dy;
        const float px = (float)w + (float)(k % 3 - 1) + dx;

        const float y0f = floorf(py);
        const float x0f = floorf(px);
        const float fy  = py - y0f;
        const float fx  = px - x0f;
        const int y0 = (int)y0f, x0 = (int)x0f;
        const int y1 = y0 + 1,   x1 = x0 + 1;

        const bool vy0 = (y0 >= 0) && (y0 < HH);
        const bool vy1 = (y1 >= 0) && (y1 < HH);
        const bool vx0 = (x0 >= 0) && (x0 < WW);
        const bool vx1 = (x1 >= 0) && (x1 < WW);

        const int y0c = min(max(y0, 0), HH - 1);
        const int y1c = min(max(y1, 0), HH - 1);
        const int x0c = min(max(x0, 0), WW - 1);
        const int x1c = min(max(x1, 0), WW - 1);

        int4 id;
        id.x = y0c * WW + x0c;
        id.y = y0c * WW + x1c;
        id.z = y1c * WW + x0c;
        id.w = y1c * WW + x1c;

        float4 wv;
        wv.x = (1.0f - fy) * (1.0f - fx) * (float)(vy0 && vx0);
        wv.y = (1.0f - fy) * fx          * (float)(vy0 && vx1);
        wv.z = fy * (1.0f - fx)          * (float)(vy1 && vx0);
        wv.w = fy * fx                   * (float)(vy1 && vx1);

        s_idx[i] = id;
        s_w[i]   = wv;
    }
    __syncthreads();

    const int p    = tid & 63;            // pixel within tile
    const int cofs = tid >> 6;            // 0..3 channel phase
    const int pix  = tileBase + p;

    for (int c = cofs; c < CH; c += 4) {
        const float* __restrict__ xp = x + ((size_t)b * CH + c) * HWSZ;
        const float* __restrict__ wk = &s_wd[c * KTAP];

        // Pull next channel plane toward the caches (global_prefetch_b8)
        __builtin_prefetch(xp + 4 * (size_t)HWSZ, 0, 1);

        float acc = 0.0f;
#pragma unroll
        for (int k = 0; k < KTAP; ++k) {
            const int4   id = s_idx[(k << 6) + p];
            const float4 wv = s_w[(k << 6) + p];
            const float bil = xp[id.x] * wv.x + xp[id.y] * wv.y
                            + xp[id.z] * wv.z + xp[id.w] * wv.w;
            acc = fmaf(wk[k], bil, acc);
        }
        out[((size_t)b * CH + c) * HWSZ + pix] = acc;
    }
}

// ---------------------------------------------------------------------------
// Launch: kernel1 writes offsets into d_ws (1.8 MB), kernel2 consumes them.
// Inputs (setup_inputs order): x, w_off, b_off, w_def. Output f32 [B,C,H,W].
// ---------------------------------------------------------------------------
extern "C" void kernel_launch(void* const* d_in, const int* in_sizes, int n_in,
                              void* d_out, int out_size, void* d_ws, size_t ws_size,
                              hipStream_t stream)
{
    const float* x     = (const float*)d_in[0];
    const float* w_off = (const float*)d_in[1];
    const float* b_off = (const float*)d_in[2];
    const float* w_def = (const float*)d_in[3];
    float*       out   = (float*)d_out;
    float*       offb  = (float*)d_ws;   // B*18*H*W floats = 1.8 MB

    // 1568 M-tiles of 16 pixels, 4 waves (tiles) per 128-thread block
    deform_offset_wmma<<<dim3(MTOT / 16 / 4), dim3(128), 0, stream>>>(
        x, w_off, b_off, offb);

    // 49 pixel-tiles x 8 images, 256 threads (64 px * 4 ch)
    deform_apply<<<dim3(HWSZ / 64, BATCH), dim3(256), 0, stream>>>(
        x, offb, w_def, out);
}
